// STML_loss_91285234909295
// MI455X (gfx1250) — compile-verified, matrix operand order checked by reading, not run on hardware
//
#include <hip/hip_runtime.h>
#include <hip/hip_bf16.h>

typedef __attribute__((ext_vector_type(16))) _Float16 v16h;
typedef __attribute__((ext_vector_type(8)))  _Float16 v8h;
typedef __attribute__((ext_vector_type(8)))  float    v8f;

#define NPTS 4096
#define MASKW 128          // 4096 / 32
#define K_TOP 10
#define K_HALF 5

// ---------------------------------------------------------------------------
// 1) row L2 normalize -> f16
// ---------------------------------------------------------------------------
__global__ void k_l2norm_half(const float* __restrict__ X, _Float16* __restrict__ Y, int D) {
    int row = blockIdx.x;
    const float* x = X + (size_t)row * D;
    _Float16* y = Y + (size_t)row * D;
    float ss = 0.f;
    for (int j = threadIdx.x; j < D; j += blockDim.x) { float v = x[j]; ss += v * v; }
    __shared__ float red[256];
    red[threadIdx.x] = ss; __syncthreads();
    for (int s = blockDim.x / 2; s > 0; s >>= 1) {
        if (threadIdx.x < s) red[threadIdx.x] += red[threadIdx.x + s];
        __syncthreads();
    }
    float inv = 1.0f / fmaxf(sqrtf(red[0]), 1e-12f);
    for (int j = threadIdx.x; j < D; j += blockDim.x) y[j] = (_Float16)(x[j] * inv);
}

// ---------------------------------------------------------------------------
// 2) WMMA Gram: one wave -> 32x32 tile of X·Xᵀ, fused epilogue.
//    MODE 0: out = exp(-max(2-2g,0))   (W_P, SIGMA=1)
//    MODE 1: out = sqrt(max(2-2g,0))   (pairwise distance)
// A-frag (16x32, lane=M) and B-frag of Aᵀ (32x16, lane=N) share the same
// per-lane packing: lanes 0-15 hold K {0..7,16..23}, lanes 16-31 {8..15,24..31}.
// ---------------------------------------------------------------------------
__device__ __forceinline__ v16h load_frag(const _Float16* __restrict__ base,
                                          int row0, int D, int k0, int lane) {
    int half = lane >> 4;
    int m = lane & 15;
    const _Float16* p = base + (size_t)(row0 + m) * D + k0 + half * 8;
    v8h lo = *reinterpret_cast<const v8h*>(p);
    v8h hi = *reinterpret_cast<const v8h*>(p + 16);
    v16h r;
#pragma unroll
    for (int i = 0; i < 8; i++) { r[i] = lo[i]; r[8 + i] = hi[i]; }
    return r;
}

template <int MODE>
__global__ void k_gram(const _Float16* __restrict__ A, float* __restrict__ Out, int D) {
    const int N = NPTS;
    int lane = threadIdx.x;              // blockDim.x == 32 (one wave32)
    int i0 = blockIdx.y * 32;
    int j0 = blockIdx.x * 32;
    v8f c00 = {}, c01 = {}, c10 = {}, c11 = {};
    for (int k0 = 0; k0 < D; k0 += 32) {
        v16h a0 = load_frag(A, i0,      D, k0, lane);
        v16h a1 = load_frag(A, i0 + 16, D, k0, lane);
        v16h b0 = load_frag(A, j0,      D, k0, lane);
        v16h b1 = load_frag(A, j0 + 16, D, k0, lane);
        c00 = __builtin_amdgcn_wmma_f32_16x16x32_f16(false, a0, false, b0, (short)0, c00, false, false);
        c01 = __builtin_amdgcn_wmma_f32_16x16x32_f16(false, a0, false, b1, (short)0, c01, false, false);
        c10 = __builtin_amdgcn_wmma_f32_16x16x32_f16(false, a1, false, b0, (short)0, c10, false, false);
        c11 = __builtin_amdgcn_wmma_f32_16x16x32_f16(false, a1, false, b1, (short)0, c11, false, false);
    }
    int half = lane >> 4;
    int col  = lane & 15;
    auto store = [&](v8f c, int rbase, int cbase) {
#pragma unroll
        for (int r = 0; r < 8; r++) {
            int row = rbase + r + 8 * half;       // C/D layout: lanes16-31 -> M+8
            float g  = c[r];
            float d2 = fmaxf(2.0f - 2.0f * g, 0.0f);
            float val = (MODE == 0) ? __expf(-d2) : sqrtf(d2);
            Out[(size_t)row * N + (cbase + col)] = val;
        }
    };
    store(c00, i0, j0); store(c01, i0, j0 + 16);
    store(c10, i0 + 16, j0); store(c11, i0 + 16, j0 + 16);
}

// ---------------------------------------------------------------------------
// 3) per-row inverse mean of a distance matrix row
// ---------------------------------------------------------------------------
__global__ void k_row_invmean(const float* __restrict__ M, float* __restrict__ invmean) {
    int row = blockIdx.x;
    const float* x = M + (size_t)row * NPTS;
    float s = 0.f;
    for (int j = threadIdx.x; j < NPTS; j += blockDim.x) s += x[j];
    __shared__ float red[256];
    red[threadIdx.x] = s; __syncthreads();
    for (int st = blockDim.x / 2; st > 0; st >>= 1) {
        if (threadIdx.x < st) red[threadIdx.x] += red[threadIdx.x + st];
        __syncthreads();
    }
    if (threadIdx.x == 0) invmean[row] = (float)NPTS / red[0];
}

// ---------------------------------------------------------------------------
// 4) top-10 per row of W_P_copy (same_id forced to 1.0); ties -> lowest index
// ---------------------------------------------------------------------------
__global__ void k_topk(const float* __restrict__ W_P, const int* __restrict__ idx,
                       int* __restrict__ topk) {
    int i = blockIdx.x;
    __shared__ float vals[NPTS];
    __shared__ float rv[256];
    __shared__ int   ri[256];
    int myid = idx[i];
    for (int j = threadIdx.x; j < NPTS; j += blockDim.x) {
        float w = W_P[(size_t)i * NPTS + j];
        if (idx[j] == myid) w = 1.0f;
        vals[j] = w;
    }
    __syncthreads();
    for (int k = 0; k < K_TOP; k++) {
        float bv = -1.0f; int bi = NPTS;
        for (int j = threadIdx.x; j < NPTS; j += blockDim.x) {
            float v = vals[j];
            if (v > bv || (v == bv && j < bi)) { bv = v; bi = j; }
        }
        rv[threadIdx.x] = bv; ri[threadIdx.x] = bi; __syncthreads();
        for (int s = blockDim.x / 2; s > 0; s >>= 1) {
            if (threadIdx.x < s) {
                float v2 = rv[threadIdx.x + s]; int i2 = ri[threadIdx.x + s];
                if (v2 > rv[threadIdx.x] || (v2 == rv[threadIdx.x] && i2 < ri[threadIdx.x])) {
                    rv[threadIdx.x] = v2; ri[threadIdx.x] = i2;
                }
            }
            __syncthreads();
        }
        if (threadIdx.x == 0) { topk[i * K_TOP + k] = ri[0]; vals[ri[0]] = -2.0f; }
        __syncthreads();
    }
}

// ---------------------------------------------------------------------------
// 5) W_NN bitmask scatter; mutual-kNN mask V; degrees
// ---------------------------------------------------------------------------
__global__ void k_wnn(const int* __restrict__ topk, unsigned* __restrict__ WNN) {
    int i = blockIdx.x * blockDim.x + threadIdx.x;
    if (i >= NPTS) return;
    for (int k = 0; k < K_TOP; k++) {
        int j = topk[i * K_TOP + k];
        atomicOr(&WNN[(size_t)i * MASKW + (j >> 5)], 1u << (j & 31));
    }
}

__global__ void k_vmask(const unsigned* __restrict__ WNN, unsigned* __restrict__ V) {
    int p = blockIdx.x * blockDim.x + threadIdx.x;   // p = i*MASKW + w
    if (p >= NPTS * MASKW) return;
    int i = p / MASKW, w = p % MASKW;
    unsigned m = WNN[p], out = 0;
    int iw = i >> 5, ib = i & 31;
    while (m) {
        int b = __ffs(m) - 1; m &= m - 1;
        int j = w * 32 + b;
        if ((WNN[(size_t)j * MASKW + iw] >> ib) & 1u) out |= 1u << b;
    }
    V[p] = out;
}

__global__ void k_deg(const unsigned* __restrict__ V, int* __restrict__ deg) {
    int i = blockIdx.x * blockDim.x + threadIdx.x;
    if (i >= NPTS) return;
    int d = 0;
    for (int w = 0; w < MASKW; w++) d += __popc(V[(size_t)i * MASKW + w]);
    deg[i] = d;
}

// ---------------------------------------------------------------------------
// 6) W_C_hat[i,:] = mean over 5 half-neighbors h of W_C_tilda[h,:]
//    W_C_tilda[h,j] = (V[h,j] ? popcount(V_h & V_j)/max(deg_h,1) : 0)
//    WCH pre-zeroed; thread w owns j in [32w,32w+32) -> no races.
// ---------------------------------------------------------------------------
__global__ void k_wch(const unsigned* __restrict__ V, const int* __restrict__ topk,
                      const int* __restrict__ deg, float* __restrict__ WCH) {
    int i = blockIdx.x;
    int w = threadIdx.x;                    // blockDim.x == MASKW (128)
    for (int kh = 0; kh < K_HALF; kh++) {
        int h = topk[i * K_TOP + kh];
        float inv = 1.0f / (5.0f * (float)max(deg[h], 1));
        unsigned m = V[(size_t)h * MASKW + w];
        while (m) {
            int b = __ffs(m) - 1; m &= m - 1;
            int j = w * 32 + b;
            int vv = 0;
            for (int w2 = 0; w2 < MASKW; w2++)
                vv += __popc(V[(size_t)h * MASKW + w2] & V[(size_t)j * MASKW + w2]);
            WCH[(size_t)i * NPTS + j] += (float)vv * inv;
        }
    }
}

// ---------------------------------------------------------------------------
// 7) fused RC reduction for both s_f and s_g (shared weights from t_g)
// ---------------------------------------------------------------------------
__global__ void k_rc(const float* __restrict__ W_P, const float* __restrict__ WCH,
                     const float* __restrict__ Sf, const float* __restrict__ Sg,
                     const float* __restrict__ imf, const float* __restrict__ img,
                     double* __restrict__ acc) {
    size_t total = (size_t)NPTS * NPTS;
    double af = 0.0, ag = 0.0;
    for (size_t p = (size_t)blockIdx.x * blockDim.x + threadIdx.x; p < total;
         p += (size_t)gridDim.x * blockDim.x) {
        int i = (int)(p / NPTS), j = (int)(p % NPTS);
        if (i == j) continue;
        float wc = 0.5f * (WCH[p] + WCH[(size_t)j * NPTS + i]);
        float W  = 0.5f * (W_P[p] + wc);
        float posw = W, negw = 1.0f - W;
        float sdf = Sf[p] * imf[i]; float tf = fmaxf(1.0f - sdf, 0.0f);
        af += (double)(sdf * sdf * posw + tf * tf * negw);
        float sdg = Sg[p] * img[i]; float tg = fmaxf(1.0f - sdg, 0.0f);
        ag += (double)(sdg * sdg * posw + tg * tg * negw);
    }
    __shared__ double rf[256], rg[256];
    rf[threadIdx.x] = af; rg[threadIdx.x] = ag; __syncthreads();
    for (int s = blockDim.x / 2; s > 0; s >>= 1) {
        if (threadIdx.x < s) { rf[threadIdx.x] += rf[threadIdx.x + s]; rg[threadIdx.x] += rg[threadIdx.x + s]; }
        __syncthreads();
    }
    if (threadIdx.x == 0) { atomicAdd(&acc[0], rf[0]); atomicAdd(&acc[1], rg[0]); }
}

// ---------------------------------------------------------------------------
// 8) KL(p||q): q from s_f distances, p from s_g distances (one block per row)
// ---------------------------------------------------------------------------
__global__ void k_kl(const float* __restrict__ Sf, const float* __restrict__ Sg,
                     const float* __restrict__ imf, const float* __restrict__ img,
                     double* __restrict__ acc) {
    int i = blockIdx.x;
    __shared__ float ql[NPTS];
    __shared__ float pl[NPTS];
    __shared__ float r1[256], r2[256];
    __shared__ double rd[256];
    float invf = imf[i], invg = img[i];
    for (int j = threadIdx.x; j < NPTS; j += blockDim.x) {
        ql[j] = -Sf[(size_t)i * NPTS + j] * invf;
        pl[j] = -Sg[(size_t)i * NPTS + j] * invg;
    }
    __syncthreads();
    float mq = -1e30f, mp = -1e30f;
    for (int j = threadIdx.x; j < NPTS; j += blockDim.x) { mq = fmaxf(mq, ql[j]); mp = fmaxf(mp, pl[j]); }
    r1[threadIdx.x] = mq; r2[threadIdx.x] = mp; __syncthreads();
    for (int s = blockDim.x / 2; s > 0; s >>= 1) {
        if (threadIdx.x < s) {
            r1[threadIdx.x] = fmaxf(r1[threadIdx.x], r1[threadIdx.x + s]);
            r2[threadIdx.x] = fmaxf(r2[threadIdx.x], r2[threadIdx.x + s]);
        }
        __syncthreads();
    }
    mq = r1[0]; mp = r2[0]; __syncthreads();
    float sq = 0.f, sp = 0.f;
    for (int j = threadIdx.x; j < NPTS; j += blockDim.x) { sq += __expf(ql[j] - mq); sp += __expf(pl[j] - mp); }
    r1[threadIdx.x] = sq; r2[threadIdx.x] = sp; __syncthreads();
    for (int s = blockDim.x / 2; s > 0; s >>= 1) {
        if (threadIdx.x < s) { r1[threadIdx.x] += r1[threadIdx.x + s]; r2[threadIdx.x] += r2[threadIdx.x + s]; }
        __syncthreads();
    }
    float lseq = mq + __logf(r1[0]);
    float lsep = mp + __logf(r2[0]);
    __syncthreads();
    double s = 0.0;
    for (int j = threadIdx.x; j < NPTS; j += blockDim.x) {
        float logp = pl[j] - lsep;
        float logq = ql[j] - lseq;
        s += (double)(__expf(logp) * (logp - logq));
    }
    rd[threadIdx.x] = s; __syncthreads();
    for (int st = blockDim.x / 2; st > 0; st >>= 1) {
        if (threadIdx.x < st) rd[threadIdx.x] += rd[threadIdx.x + st];
        __syncthreads();
    }
    if (threadIdx.x == 0) atomicAdd(&acc[2], rd[0]);
}

// ---------------------------------------------------------------------------
// 9) finalize 3 scalars
// ---------------------------------------------------------------------------
__global__ void k_final(const double* __restrict__ acc, float* __restrict__ out) {
    if (threadIdx.x == 0 && blockIdx.x == 0) {
        double denom = (double)NPTS * (double)(NPTS - 1);
        double rc = 0.5 * (acc[0] / denom + acc[1] / denom);
        double kl = acc[2] / (double)NPTS;
        out[0] = (float)rc;
        out[1] = (float)kl;
        out[2] = (float)(rc + kl);
    }
}

// ---------------------------------------------------------------------------
extern "C" void kernel_launch(void* const* d_in, const int* in_sizes, int n_in,
                              void* d_out, int out_size, void* d_ws, size_t ws_size,
                              hipStream_t stream) {
    const float* s_f = (const float*)d_in[0];   // 4096 x 512
    const float* s_g = (const float*)d_in[1];   // 4096 x 512
    const float* t_g = (const float*)d_in[2];   // 4096 x 1024
    const int*   idx = (const int*)d_in[3];     // 4096
    float* out = (float*)d_out;

    char* w = (char*)d_ws;
    size_t off = 0;
    auto alloc = [&](size_t bytes) -> void* {
        void* p = w + off;
        off = (off + bytes + 255) & ~(size_t)255;
        return p;
    };
    _Float16* t_h  = (_Float16*)alloc((size_t)NPTS * 1024 * 2);
    _Float16* sf_h = (_Float16*)alloc((size_t)NPTS * 512 * 2);
    _Float16* sg_h = (_Float16*)alloc((size_t)NPTS * 512 * 2);
    float* W_P = (float*)alloc((size_t)NPTS * NPTS * 4);
    float* Sf  = (float*)alloc((size_t)NPTS * NPTS * 4);
    float* Sg  = (float*)alloc((size_t)NPTS * NPTS * 4);
    float* WCH = (float*)alloc((size_t)NPTS * NPTS * 4);
    unsigned* WNN = (unsigned*)alloc((size_t)NPTS * MASKW * 4);
    unsigned* V   = (unsigned*)alloc((size_t)NPTS * MASKW * 4);
    int* topk = (int*)alloc((size_t)NPTS * K_TOP * 4);
    int* deg  = (int*)alloc((size_t)NPTS * 4);
    float* imf = (float*)alloc((size_t)NPTS * 4);
    float* img = (float*)alloc((size_t)NPTS * 4);
    double* acc = (double*)alloc(4 * sizeof(double));

    // zero what must start at zero each call
    hipMemsetAsync(WCH, 0, (size_t)NPTS * NPTS * 4, stream);
    hipMemsetAsync(WNN, 0, (size_t)NPTS * MASKW * 4, stream);
    hipMemsetAsync(acc, 0, 4 * sizeof(double), stream);

    // normalize
    k_l2norm_half<<<NPTS, 256, 0, stream>>>(t_g, t_h, 1024);
    k_l2norm_half<<<NPTS, 256, 0, stream>>>(s_f, sf_h, 512);
    k_l2norm_half<<<NPTS, 256, 0, stream>>>(s_g, sg_h, 512);

    // WMMA Grams with fused epilogues
    dim3 gg(NPTS / 32, NPTS / 32);
    k_gram<0><<<gg, 32, 0, stream>>>(t_h, W_P, 1024);   // W_P = exp(-d^2)
    k_gram<1><<<gg, 32, 0, stream>>>(sf_h, Sf, 512);    // raw distances
    k_gram<1><<<gg, 32, 0, stream>>>(sg_h, Sg, 512);

    // row inverse means
    k_row_invmean<<<NPTS, 256, 0, stream>>>(Sf, imf);
    k_row_invmean<<<NPTS, 256, 0, stream>>>(Sg, img);

    // graph weights
    k_topk<<<NPTS, 256, 0, stream>>>(W_P, idx, topk);
    k_wnn<<<(NPTS + 255) / 256, 256, 0, stream>>>(topk, WNN);
    k_vmask<<<(NPTS * MASKW + 255) / 256, 256, 0, stream>>>(WNN, V);
    k_deg<<<(NPTS + 255) / 256, 256, 0, stream>>>(V, deg);
    k_wch<<<NPTS, MASKW, 0, stream>>>(V, topk, deg, WCH);

    // losses
    k_rc<<<4096, 256, 0, stream>>>(W_P, WCH, Sf, Sg, imf, img, acc);
    k_kl<<<NPTS, 256, 0, stream>>>(Sf, Sg, imf, img, acc);
    k_final<<<1, 32, 0, stream>>>(acc, out);
}
